// GCN_62637803045081
// MI455X (gfx1250) — matrix operand order repaired and checked by reading, compiled, not run on hardware
//
#include <hip/hip_runtime.h>
#include <stdint.h>

// ---------------------------------------------------------------------------
// GCN 2-layer forward for MI455X (gfx1250), wave32 + WMMA.
//   layer: h = X@W ; agg[d] += h[s]*dinv[s]*dinv[d] over edges ;
//          out = relu(agg + h*dinv^2 (self loop) + b)
// ---------------------------------------------------------------------------

typedef __attribute__((ext_vector_type(16))) _Float16 v16h;
typedef __attribute__((ext_vector_type(8)))  float    v8f;

#define GCN_N 100000

// ---------------- utility kernels -----------------------------------------

__global__ void gcn_fill(float* __restrict__ p, float v, long long n) {
    long long i = (long long)blockIdx.x * blockDim.x + threadIdx.x;
    if (i < n) p[i] = v;
}

__global__ void gcn_deg_accum(const long long* __restrict__ dst, float* __restrict__ deg,
                              long long E) {
    long long e = (long long)blockIdx.x * blockDim.x + threadIdx.x;
    if (e < E) atomicAdd(&deg[(int)dst[e]], 1.0f);
}

__global__ void gcn_rsqrt(float* __restrict__ deg, int n) {
    int i = blockIdx.x * blockDim.x + threadIdx.x;
    if (i < n) {
        float d = deg[i];
        deg[i] = (d > 0.0f) ? rsqrtf(d) : 0.0f;   // in place: deg -> dinv
    }
}

// ---------------- WMMA GEMM: H[M,64] = X[M,K] @ W[K,64] --------------------
// One wave computes a 16-row x 64-col strip: 4 accumulators of 16x16,
// K/32 fully-unrolled wmma steps each.
// W is staged in LDS transposed + pre-converted to f16 (sWt[n][k], padded
// row stride) so each B fragment is one contiguous, 32B-aligned v16h read.

template <int K>
__global__ void gcn_gemm_wmma(const float* __restrict__ X, const float* __restrict__ W,
                              float* __restrict__ H, int M) {
    constexpr int LDK = K + 16;                       // halfs; keeps rows 32B-aligned
    __shared__ __align__(32) _Float16 sWt[64 * LDK];

    // transpose + convert: W[k*64+n] (fp32 row-major) -> sWt[n*LDK + k] (f16)
    for (int i = threadIdx.x; i < K * 64; i += blockDim.x) {
        int k = i >> 6, n = i & 63;
        sWt[n * LDK + k] = (_Float16)W[i];
    }
    __syncthreads();

    const int lane = threadIdx.x & 31;
    const int wave = threadIdx.x >> 5;
    const int tile = blockIdx.x * (blockDim.x >> 5) + wave;   // 16-row tile id
    const int mrow = tile * 16;
    if (mrow >= M) return;           // wave-uniform: EXEC stays all-1 for WMMA

    const int m  = lane & 15;        // M row (A), N col (B/C/D)
    const int hi = lane >> 4;        // lane half selector

    v8f acc[4] = {v8f{}, v8f{}, v8f{}, v8f{}};

    const float* xrow = X + (size_t)(mrow + m) * K;

    #pragma unroll
    for (int kt = 0; kt < K / 32; ++kt) {
        const int kbase = kt * 32;

        // A fragment: 16x32 f16, layout: half j -> K = hi*16 + j
        v16h a;
        #pragma unroll
        for (int g = 0; g < 2; ++g) {
            const float4* p = (const float4*)(xrow + kbase + hi * 16 + g * 8);
            float4 f0 = p[0], f1 = p[1];
            a[g * 8 + 0] = (_Float16)f0.x;  a[g * 8 + 1] = (_Float16)f0.y;
            a[g * 8 + 2] = (_Float16)f0.z;  a[g * 8 + 3] = (_Float16)f0.w;
            a[g * 8 + 4] = (_Float16)f1.x;  a[g * 8 + 5] = (_Float16)f1.y;
            a[g * 8 + 6] = (_Float16)f1.z;  a[g * 8 + 7] = (_Float16)f1.w;
        }

        #pragma unroll
        for (int nt = 0; nt < 4; ++nt) {
            // B fragment: 32x16 f16; lane col = m, halves j -> K = hi*16 + j
            // contiguous 32B aligned read from transposed LDS copy
            v16h b = *(const v16h*)&sWt[(size_t)(nt * 16 + m) * LDK + kbase + hi * 16];

            acc[nt] = __builtin_amdgcn_wmma_f32_16x16x32_f16(
                false, a, false, b, (short)0, acc[nt], false, false);
        }
    }

    // C/D layout: VGPR r holds row M = r + 8*hi, column = m
    #pragma unroll
    for (int nt = 0; nt < 4; ++nt)
        #pragma unroll
        for (int r = 0; r < 8; ++r)
            H[(size_t)(mrow + r + 8 * hi) * 64 + nt * 16 + m] = acc[nt][r];
}

// ---------------- edge scatter: agg[dst] += h[src]*dinv[src]*dinv[dst] -----
// 64 threads per edge, one column each; e is wave-uniform so index/norm
// loads scalarize.  Gather of h[src] is a coalesced 256B read (L2-resident);
// scatter is f32 L2 atomics.

__global__ void gcn_scatter(const long long* __restrict__ srcIdx,
                            const long long* __restrict__ dstIdx,
                            const float* __restrict__ dinv,
                            const float* __restrict__ H,
                            float* __restrict__ AGG, long long E) {
    long long idx = (long long)blockIdx.x * blockDim.x + threadIdx.x;
    long long e = idx >> 6;
    int c = (int)(idx & 63);
    if (e >= E) return;
    int s = (int)srcIdx[e];
    int d = (int)dstIdx[e];
    float nrm = dinv[s] * dinv[d];
    float val = H[(size_t)s * 64 + c] * nrm;
    atomicAdd(&AGG[(size_t)d * 64 + c], val);
}

// ---------------- epilogue: out = relu(agg + h*dinv^2 + bias) --------------

__global__ void gcn_epilogue(const float* __restrict__ AGG, const float* __restrict__ H,
                             const float* __restrict__ dinv, const float* __restrict__ bias,
                             float* __restrict__ OUT, int n) {
    long long idx = (long long)blockIdx.x * blockDim.x + threadIdx.x;
    if (idx >= (long long)n * 64) return;
    int i = (int)(idx >> 6);
    int c = (int)(idx & 63);
    float di = dinv[i];
    float v = AGG[idx] + H[idx] * di * di + bias[c];
    OUT[idx] = v > 0.0f ? v : 0.0f;
}

// ---------------------------------------------------------------------------

extern "C" void kernel_launch(void* const* d_in, const int* in_sizes, int n_in,
                              void* d_out, int out_size, void* d_ws, size_t ws_size,
                              hipStream_t stream) {
    (void)n_in; (void)out_size; (void)ws_size;

    const float*     x  = (const float*)d_in[0];
    const long long* ei = (const long long*)d_in[1];   // [2, E] int64
    const float*     W1 = (const float*)d_in[2];
    const float*     b1 = (const float*)d_in[3];
    const float*     W2 = (const float*)d_in[4];
    const float*     b2 = (const float*)d_in[5];
    float*           out = (float*)d_out;

    const int       N = GCN_N;
    const long long E = (long long)in_sizes[1] / 2;
    const long long NF = (long long)N * 64;

    // workspace layout (floats): dinv[N] | bufA[N*64] | bufB[N*64] | bufC[N*64]
    float* dinv = (float*)d_ws;
    float* bufA = dinv + N;        // h (linear transform result)
    float* bufB = bufA + NF;       // agg (atomic scatter target)
    float* bufC = bufB + NF;       // a1 (layer-1 activation)

    const int T = 256;
    const int blkN   = (N + T - 1) / T;
    const int blkE   = (int)((E + T - 1) / T);
    const int blkE64 = (int)((E * 64 + T - 1) / T);
    const int blkNF  = (int)((NF + T - 1) / T);
    const int tiles  = (N + 15) / 16;                // 6250
    const int blkG   = (tiles + (T / 32) - 1) / (T / 32);

    const long long* srcI = ei;
    const long long* dstI = ei + E;

    // --- normalization: deg = 1 (self loop) + scatter(ones); dinv = rsqrt ---
    gcn_fill<<<blkN, T, 0, stream>>>(dinv, 1.0f, N);
    gcn_deg_accum<<<blkE, T, 0, stream>>>(dstI, dinv, E);
    gcn_rsqrt<<<blkN, T, 0, stream>>>(dinv, N);

    // --- layer 1 ---
    gcn_gemm_wmma<128><<<blkG, T, 0, stream>>>(x, W1, bufA, N);
    gcn_fill<<<blkNF, T, 0, stream>>>(bufB, 0.0f, NF);
    gcn_scatter<<<blkE64, T, 0, stream>>>(srcI, dstI, dinv, bufA, bufB, E);
    gcn_epilogue<<<blkNF, T, 0, stream>>>(bufB, bufA, dinv, b1, bufC, N);

    // --- layer 2 ---
    gcn_gemm_wmma<64><<<blkG, T, 0, stream>>>(bufC, W2, bufA, N);
    gcn_fill<<<blkNF, T, 0, stream>>>(bufB, 0.0f, NF);
    gcn_scatter<<<blkE64, T, 0, stream>>>(srcI, dstI, dinv, bufA, bufB, E);
    gcn_epilogue<<<blkNF, T, 0, stream>>>(bufB, bufA, dinv, b2, out, N);
}
